// GraphModel_12017318494508
// MI455X (gfx1250) — matrix operand order repaired and checked by reading, not compiled, not run on hardware
//
#include <hip/hip_runtime.h>
#include <math.h>

// GATv2 x2 + global-attention pooling for MI455X (gfx1250, wave32, WMMA).
// GEMMs: one-shot f32->f16 prep (X row-major, W transposed to [m][k]) so the
// WMMA inner loop is pure b128 loads + v_wmma_f32_16x16x32_f16 (f32 accum).
// Edge softmax: float-keyed atomicMax + global_atomic_add_f32 scatter.

#define SLOPE 0.2f

typedef __attribute__((ext_vector_type(16))) _Float16 v16h;
typedef __attribute__((ext_vector_type(8)))  _Float16 v8h;
typedef __attribute__((ext_vector_type(8)))  float    v8f;

// ---------------------------------------------------------------------------
// One-shot precision prep kernels (tiny vs. the edge phase traffic).
// ---------------------------------------------------------------------------
__global__ void f32_to_f16(const float* __restrict__ in,
                           _Float16* __restrict__ out, int n)
{
  int i = blockIdx.x * blockDim.x + threadIdx.x;
  if (i < n) out[i] = (_Float16)in[i];
}

// Wt[c][r] = (f16) W[r][c]   (W is [k,m] row-major -> Wt is [m,k] row-major)
__global__ void transpose_to_f16(const float* __restrict__ W,
                                 _Float16* __restrict__ Wt, int k, int m)
{
  int i = blockIdx.x * blockDim.x + threadIdx.x;   // i = c*k + r
  if (i >= k * m) return;
  int c = i / k, r = i % k;
  Wt[i] = (_Float16)W[(size_t)r * m + c];
}

// ---------------------------------------------------------------------------
// Y[n,m] = X[n,k] @ W[k,m] + bias[m] with f16 inputs (Xh row-major, Wt = W^T).
// One wave -> one 16x16 C tile; blockDim (32,4) covers 4 col-tiles.
// A (16x32 f16, ISA 7.12.2): lane l row=l%16; vgpr j: K = (j>>2)*16 + (l/16)*8 + 2*(j&3)+{0,1}
//   -> per lane: f16 runs [k0+half*8, +8) and [k0+16+half*8, +8)  (2x b128)
// B (32x16 f16): lane l col=l%16; same K order as A
//   -> per lane: f16 runs [k0+half*8, +8) and [k0+16+half*8, +8) of Wt[col] (2x b128)
// C (16x16 f32): lane l col=l%16; vgpr j: row = (l/16)*8 + j
// ---------------------------------------------------------------------------
template <int K>
__global__ void __launch_bounds__(128) gemm_bias_wmma(
    const _Float16* __restrict__ Xh, const _Float16* __restrict__ Wt,
    const float* __restrict__ bias, float* __restrict__ Y,
    int n, int k, int m)
{
  const int lane = threadIdx.x;                    // 0..31
  const int tm   = blockIdx.x;                     // 16-row tile
  const int tn   = blockIdx.y * 4 + threadIdx.y;   // 16-col tile
  const int half = lane >> 4;
  const int lid  = lane & 15;
  const int arow = min(tm * 16 + lid, n - 1);
  const int bcol = tn * 16 + lid;
  const int kk   = (K > 0) ? K : k;

  const _Float16* ap = Xh + (size_t)arow * kk;
  const _Float16* bp = Wt + (size_t)bcol * kk;

  v8f acc = {};
#pragma unroll
  for (int k0 = 0; k0 < kk; k0 += 32) {
    const v8h a0 = *(const v8h*)(ap + k0 + half * 8);
    const v8h a1 = *(const v8h*)(ap + k0 + half * 8 + 16);
    const v8h b0 = *(const v8h*)(bp + k0 + half * 8);
    const v8h b1 = *(const v8h*)(bp + k0 + half * 8 + 16);
    const v16h a = __builtin_shufflevector(a0, a1, 0, 1, 2, 3, 4, 5, 6, 7,
                                           8, 9, 10, 11, 12, 13, 14, 15);
    const v16h b = __builtin_shufflevector(b0, b1, 0, 1, 2, 3, 4, 5, 6, 7,
                                           8, 9, 10, 11, 12, 13, 14, 15);
    // (neg_a, A, neg_b, B, c_mod, C, reuse_a, reuse_b)
    acc = __builtin_amdgcn_wmma_f32_16x16x32_f16(false, a, false, b,
                                                 (short)0, acc, false, false);
  }

  const float bv = bias[bcol];
  if (tm * 16 + 15 < n) {              // full tile: unguarded stores
#pragma unroll
    for (int j = 0; j < 8; ++j)
      Y[(size_t)(tm * 16 + half * 8 + j) * m + bcol] = acc[j] + bv;
  } else {
#pragma unroll
    for (int j = 0; j < 8; ++j) {
      const int r = tm * 16 + half * 8 + j;
      if (r < n) Y[(size_t)r * m + bcol] = acc[j] + bv;
    }
  }
}

// ---------------------------------------------------------------------------
// Per-edge GATv2 logits: logits[e][h] = attn[h,:] . leaky_relu(fs[src]+fd[dst])
// One wave per edge; H = 4 heads.
// ---------------------------------------------------------------------------
__global__ void __launch_bounds__(256) edge_logits(
    const float* __restrict__ fs, const float* __restrict__ fd,
    const float* __restrict__ attn,              // [4, D]
    const int* __restrict__ src, const int* __restrict__ dst,
    float* __restrict__ logits, int E, int D)
{
  const int e    = (blockIdx.x * blockDim.x + threadIdx.x) >> 5;
  const int lane = threadIdx.x & 31;
  if (e >= E) return;
  const int s = src[e], d = dst[e];
  const float* ps = fs + (size_t)s * 4 * D;
  const float* pd = fd + (size_t)d * 4 * D;
#pragma unroll
  for (int h = 0; h < 4; ++h) {
    float sum = 0.f;
    for (int i = lane; i < D; i += 32) {
      float v = ps[h * D + i] + pd[h * D + i];
      v = (v > 0.f) ? v : v * SLOPE;
      sum += v * attn[h * D + i];
    }
#pragma unroll
    for (int off = 16; off; off >>= 1) sum += __shfl_xor(sum, off, 32);
    if (lane == 0) logits[(size_t)e * 4 + h] = sum;
  }
}

// Monotone float<->uint key so unsigned atomicMax == float max.
__device__ __forceinline__ unsigned f2key(float f) {
  unsigned b = __float_as_uint(f);
  return (b & 0x80000000u) ? ~b : (b | 0x80000000u);
}
__device__ __forceinline__ float key2f(unsigned kk) {
  return (kk & 0x80000000u) ? __uint_as_float(kk & 0x7FFFFFFFu)
                            : __uint_as_float(~kk);
}

__global__ void edge_segmax(const float* __restrict__ logits,
                            const int* __restrict__ dst,
                            unsigned* __restrict__ mkey, int E)
{
  int i = blockIdx.x * blockDim.x + threadIdx.x;   // over E*4
  if (i >= E * 4) return;
  int e = i >> 2, h = i & 3;
  atomicMax(&mkey[(size_t)dst[e] * 4 + h], f2key(logits[i]));
}

__global__ void edge_exp(const float* __restrict__ logits,
                         const int* __restrict__ dst,
                         const unsigned* __restrict__ mkey,
                         float* __restrict__ aex, float* __restrict__ ssum, int E)
{
  int i = blockIdx.x * blockDim.x + threadIdx.x;   // over E*4
  if (i >= E * 4) return;
  int e = i >> 2, h = i & 3;
  const size_t seg = (size_t)dst[e] * 4 + h;
  float v = expf(logits[i] - key2f(mkey[seg]));
  aex[i] = v;
  atomicAdd(&ssum[seg], v);
}

// agg[dst][h][:] += (a/s) * fs[src][h][:]   -- one wave per edge
__global__ void __launch_bounds__(256) edge_aggregate(
    const float* __restrict__ fs, const float* __restrict__ aex,
    const float* __restrict__ ssum,
    const int* __restrict__ src, const int* __restrict__ dst,
    float* __restrict__ agg, int E, int D)
{
  const int e    = (blockIdx.x * blockDim.x + threadIdx.x) >> 5;
  const int lane = threadIdx.x & 31;
  if (e >= E) return;
  const int s = src[e], d = dst[e];
  const float* ps = fs + (size_t)s * 4 * D;
  float* pa = agg + (size_t)d * 4 * D;
  for (int i = lane; i < 4 * D; i += 32) {
    const int h = i / D;
    const float w = aex[(size_t)e * 4 + h] / ssum[(size_t)d * 4 + h];
    atomicAdd(&pa[i], w * ps[i]);
  }
}

// MaxPool over the 4 heads: out[n][d] = max_h agg[n][h][d]
__global__ void head_max(const float* __restrict__ agg,
                         float* __restrict__ out, int N, int D)
{
  int i = blockIdx.x * blockDim.x + threadIdx.x;   // over N*D
  if (i >= N * D) return;
  int nidx = i / D, d = i % D;
  const float* p = agg + (size_t)nidx * 4 * D + d;
  out[i] = fmaxf(fmaxf(p[0], p[D]), fmaxf(p[2 * D], p[3 * D]));
}

// gate[n] = h2[n,:] . gate_w + gate_b   -- one wave per node
__global__ void __launch_bounds__(256) gate_kernel(
    const float* __restrict__ h2, const float* __restrict__ gw,
    const float* __restrict__ gbp, float* __restrict__ gate, int N, int D)
{
  const int nidx = (blockIdx.x * blockDim.x + threadIdx.x) >> 5;
  const int lane = threadIdx.x & 31;
  if (nidx >= N) return;
  const float* p = h2 + (size_t)nidx * D;
  float sum = 0.f;
  for (int i = lane; i < D; i += 32) sum += p[i] * gw[i];
#pragma unroll
  for (int off = 16; off; off >>= 1) sum += __shfl_xor(sum, off, 32);
  if (lane == 0) gate[nidx] = sum + gbp[0];
}

// Global attention pooling per graph (nodes contiguous per graph block).
// blockDim.x == D (128); thread t owns output dim t.
__global__ void __launch_bounds__(128) pool_kernel(
    const float* __restrict__ h2, const float* __restrict__ gate,
    float* __restrict__ out, int npg, int D)
{
  const int g = blockIdx.x;
  const int t = threadIdx.x;
  const float* gg = gate + (size_t)g * npg;
  __shared__ float red[128];

  float mx = -3.402823466e+38f;
  for (int i = t; i < npg; i += blockDim.x) mx = fmaxf(mx, gg[i]);
  red[t] = mx; __syncthreads();
  for (int sft = 64; sft; sft >>= 1) {
    if (t < sft) red[t] = fmaxf(red[t], red[t + sft]);
    __syncthreads();
  }
  mx = red[0]; __syncthreads();

  float sm = 0.f;
  for (int i = t; i < npg; i += blockDim.x) sm += expf(gg[i] - mx);
  red[t] = sm; __syncthreads();
  for (int sft = 64; sft; sft >>= 1) {
    if (t < sft) red[t] += red[t + sft];
    __syncthreads();
  }
  sm = red[0]; __syncthreads();

  const float* hb = h2 + (size_t)g * npg * D;
  float acc = 0.f;
  for (int i = 0; i < npg; ++i) acc += expf(gg[i] - mx) * hb[(size_t)i * D + t];
  out[(size_t)g * D + t] = acc / sm;
}

// ---------------------------------------------------------------------------
extern "C" void kernel_launch(void* const* d_in, const int* in_sizes, int n_in,
                              void* d_out, int out_size, void* d_ws, size_t ws_size,
                              hipStream_t stream)
{
  const float* x    = (const float*)d_in[0];
  const int*   esrc = (const int*)d_in[1];
  const int*   edst = (const int*)d_in[2];
  // d_in[3] = node_graph: graphs are contiguous equal blocks -> implicit
  const float* Wl1  = (const float*)d_in[4];
  const float* bl1  = (const float*)d_in[5];
  const float* Wr1  = (const float*)d_in[6];
  const float* br1  = (const float*)d_in[7];
  const float* at1  = (const float*)d_in[8];
  const float* Wl2  = (const float*)d_in[9];
  const float* bl2  = (const float*)d_in[10];
  const float* Wr2  = (const float*)d_in[11];
  const float* br2  = (const float*)d_in[12];
  const float* at2  = (const float*)d_in[13];
  const float* gw   = (const float*)d_in[14];
  const float* gbp  = (const float*)d_in[15];

  const int E    = in_sizes[1];
  const int N    = in_sizes[3];
  const int IN   = in_sizes[0] / N;          // 128
  const int HID0 = in_sizes[8] / 4;          // 64
  const int HID1 = in_sizes[13] / 4;         // 128
  const int G    = out_size / HID1;          // 50

  // ---- workspace layout; big fp32 buffers sized for layer 2 (4*HID1)
  float* ws    = (float*)d_ws;
  float* fs    = ws;
  float* fd    = fs  + (size_t)N * 4 * HID1;
  float* agg   = fd  + (size_t)N * 4 * HID1;
  float* h1    = agg + (size_t)N * 4 * HID1;
  float* h2    = h1  + (size_t)N * HID0;
  float* logi  = h2  + (size_t)N * HID1;
  float* aex   = logi + (size_t)E * 4;
  unsigned* mk = (unsigned*)(aex + (size_t)E * 4);
  float* ssum  = (float*)(mk + (size_t)N * 4);
  float* gate  = ssum + (size_t)N * 4;
  // f16 staging (after fp32 region; offsets stay 16B-aligned)
  const size_t kin_max = (size_t)(IN > HID0 ? IN : HID0);
  const size_t wt_max  = (size_t)(IN * 4 * HID0 > HID0 * 4 * HID1
                                  ? IN * 4 * HID0 : HID0 * 4 * HID1);
  _Float16* Xh  = (_Float16*)(gate + N);
  _Float16* WtL = Xh  + (size_t)N * kin_max;
  _Float16* WtR = WtL + wt_max;
  const size_t need = (size_t)((char*)(WtR + wt_max) - (char*)d_ws);
  if (need > ws_size) return;

  const int ew_blocks = (int)(((long long)E * 32 + 255) / 256);  // wave/edge
  const int eh_blocks = (E * 4 + 255) / 256;                     // thread/(edge,head)

  auto launch_gemm = [&](const _Float16* Xh_, const _Float16* Wt_,
                         const float* bias, float* Y, int k, int m) {
    dim3 ggrid((N + 15) / 16, m / 64);
    dim3 gblk(32, 4);
    if (k == 128)
      gemm_bias_wmma<128><<<ggrid, gblk, 0, stream>>>(Xh_, Wt_, bias, Y, N, k, m);
    else if (k == 64)
      gemm_bias_wmma<64><<<ggrid, gblk, 0, stream>>>(Xh_, Wt_, bias, Y, N, k, m);
    else
      gemm_bias_wmma<0><<<ggrid, gblk, 0, stream>>>(Xh_, Wt_, bias, Y, N, k, m);
  };

  auto run_layer = [&](const float* Xin, int k, int D,
                       const float* Wl, const float* bl,
                       const float* Wr, const float* br,
                       const float* attn, float* hout) {
    const int m = 4 * D;
    f32_to_f16<<<(N * k + 255) / 256, 256, 0, stream>>>(Xin, Xh, N * k);
    transpose_to_f16<<<(k * m + 255) / 256, 256, 0, stream>>>(Wl, WtL, k, m);
    transpose_to_f16<<<(k * m + 255) / 256, 256, 0, stream>>>(Wr, WtR, k, m);
    launch_gemm(Xh, WtL, bl, fs, k, m);
    launch_gemm(Xh, WtR, br, fd, k, m);
    edge_logits<<<ew_blocks, 256, 0, stream>>>(fs, fd, attn, esrc, edst, logi, E, D);
    hipMemsetAsync(mk,   0, (size_t)N * 4 * sizeof(unsigned), stream);
    hipMemsetAsync(ssum, 0, (size_t)N * 4 * sizeof(float), stream);
    hipMemsetAsync(agg,  0, (size_t)N * 4 * D * sizeof(float), stream);
    edge_segmax<<<eh_blocks, 256, 0, stream>>>(logi, edst, mk, E);
    edge_exp<<<eh_blocks, 256, 0, stream>>>(logi, edst, mk, aex, ssum, E);
    edge_aggregate<<<ew_blocks, 256, 0, stream>>>(fs, aex, ssum, esrc, edst, agg, E, D);
    head_max<<<(N * D + 255) / 256, 256, 0, stream>>>(agg, hout, N, D);
  };

  run_layer(x,  IN,   HID0, Wl1, bl1, Wr1, br1, at1, h1);
  run_layer(h1, HID0, HID1, Wl2, bl2, Wr2, br2, at2, h2);

  gate_kernel<<<(int)(((long long)N * 32 + 255) / 256), 256, 0, stream>>>(
      h2, gw, gbp, gate, N, HID1);
  pool_kernel<<<G, HID1, 0, stream>>>(h2, gate, (float*)d_out, N / G, HID1);
}
